// HandsRAT_13700945674611
// MI455X (gfx1250) — matrix operand order repaired and checked by reading, not compiled, not run on hardware
//
#include <hip/hip_runtime.h>

// ---------------------------------------------------------------------------
// Problem constants
// ---------------------------------------------------------------------------
#define BB 32
#define TT 512
#define NN 21
#define CC 3
#define HH 128
#define OO 256
#define NH (NN * HH)        // 2688
#define FF (BB * TT)        // 16384 frames
#define ALPHA 0.2f

typedef __attribute__((ext_vector_type(16))) __bf16 v16bf;
typedef __attribute__((ext_vector_type(8)))  __bf16 v8bf;
typedef __attribute__((ext_vector_type(8)))  float  v8f;

// ---------------------------------------------------------------------------
// WMMA fragment helpers (CDNA5 16x16x32 bf16 layouts, ISA 7.12.2)
// A (16x32, MxK): lane L -> row m=L&15, kb=(L>>4)*8; elems 0..7 = K[kb..kb+7],
//                 elems 8..15 = K[kb+16..kb+23]
// B (32x16, KxN): lane L -> col n=L&15, koff=(L>>4)*8; elems 0..7 = K[koff..koff+7],
//                 elems 8..15 = K[koff+16..koff+23]
// C/D (16x16 f32): lane L -> col n=L&15; vgpr r -> row m = r + 8*(L>>4)
// ---------------------------------------------------------------------------
static __device__ inline v16bf load_a_frag(const __bf16* rowPtr, int kb) {
  v8bf lo = *(const v8bf*)(rowPtr + kb);        // K = kb .. kb+7
  v8bf hi = *(const v8bf*)(rowPtr + kb + 16);   // K = kb+16 .. kb+23
  v16bf r;
#pragma unroll
  for (int i = 0; i < 8; ++i) { r[i] = lo[i]; r[i + 8] = hi[i]; }
  return r;
}

static __device__ inline v16bf load_b_frag(const __bf16* p) {
  v8bf lo = *(const v8bf*)(p);
  v8bf hi = *(const v8bf*)(p + 16);
  v16bf r;
#pragma unroll
  for (int i = 0; i < 8; ++i) { r[i] = lo[i]; r[i + 8] = hi[i]; }
  return r;
}

static __device__ inline v8f wmma_bf16(v16bf a, v16bf b, v8f c) {
  return __builtin_amdgcn_wmma_f32_16x16x32_bf16(
      /*neg_a=*/false, a, /*neg_b=*/false, b,
      /*c_mod=*/(short)0, c, /*reuse_a=*/false, /*reuse_b=*/false);
}

// ---------------------------------------------------------------------------
// Kernel 1: repack conv_w [O][NH][3] -> wk[tap][O][NH] bf16 ; proj_w -> bf16;
//           zero the padding row used for out-of-range conv taps.
// ---------------------------------------------------------------------------
__global__ void convert_weights(const float* __restrict__ conv_w,
                                const float* __restrict__ proj_w,
                                __bf16* __restrict__ wk,
                                __bf16* __restrict__ pw,
                                __bf16* __restrict__ zrow) {
  const int WKN = 3 * OO * NH;          // 2064384
  const int PWN = OO * OO;              // 65536
  int idx = blockIdx.x * blockDim.x + threadIdx.x;
  if (idx < WKN) {
    int tap = idx / (OO * NH);
    int rem = idx - tap * (OO * NH);
    int o   = rem / NH;
    int i   = rem - o * NH;
    wk[idx] = (__bf16)conv_w[(size_t)o * (NH * 3) + (size_t)i * 3 + tap];
  } else if (idx < WKN + PWN) {
    int j = idx - WKN;
    pw[j] = (__bf16)proj_w[j];
  } else if (idx < WKN + PWN + NH) {
    zrow[idx - WKN - PWN] = (__bf16)0.0f;   // ws is poisoned; must zero
  }
}

// ---------------------------------------------------------------------------
// Kernel 2: GAT per frame. One wave (32 lanes) per frame, 4 frames per block.
// Each lane owns 4 hidden channels (128 / 32). Wh kept in a per-wave LDS
// slice (each lane only touches its own channels -> no cross-wave sync).
// s1/s2 reduced with wave32 butterfly shuffles.
// ---------------------------------------------------------------------------
__global__ __launch_bounds__(128) void gat_kernel(
    const float* __restrict__ x, const float* __restrict__ adj,
    const float* __restrict__ W, const float* __restrict__ a,
    const float* __restrict__ gbias, __bf16* __restrict__ hp) {
  __shared__ float s_adj[NN * NN];
  __shared__ float s_wh[4][NH];    // 4 waves * 2688 f32 = 43008 B

  const int lane = threadIdx.x & 31;
  const int wave = threadIdx.x >> 5;
  const int f = blockIdx.x * 4 + wave;

  for (int i = threadIdx.x; i < NN * NN; i += 128) s_adj[i] = adj[i];
  __syncthreads();

  const int h0 = lane * 4;
  float Wr[CC][4], a1r[4], a2r[4], br[4];
#pragma unroll
  for (int c = 0; c < CC; ++c)
#pragma unroll
    for (int q = 0; q < 4; ++q) Wr[c][q] = W[c * HH + h0 + q];
#pragma unroll
  for (int q = 0; q < 4; ++q) {
    a1r[q] = a[h0 + q];
    a2r[q] = a[HH + h0 + q];
    br[q]  = gbias[h0 + q];
  }

  const float* xf = x + (size_t)f * (NN * CC);
  float* wh = &s_wh[wave][0];
  float s1[NN], s2[NN];

  for (int n = 0; n < NN; ++n) {
    float x0 = xf[n * 3 + 0], x1 = xf[n * 3 + 1], x2 = xf[n * 3 + 2];
    float p1 = 0.f, p2 = 0.f;
#pragma unroll
    for (int q = 0; q < 4; ++q) {
      float v = x0 * Wr[0][q] + x1 * Wr[1][q] + x2 * Wr[2][q];
      wh[n * HH + h0 + q] = v;
      p1 += v * a1r[q];
      p2 += v * a2r[q];
    }
#pragma unroll
    for (int off = 16; off > 0; off >>= 1) {   // wave32 butterfly
      p1 += __shfl_xor(p1, off, 32);
      p2 += __shfl_xor(p2, off, 32);
    }
    s1[n] = p1;
    s2[n] = p2;
  }

  __bf16* hpf = hp + (size_t)f * NH;
  for (int n = 0; n < NN; ++n) {
    float er[NN];
    float mx = -3.4e38f;
#pragma unroll
    for (int j = 0; j < NN; ++j) {
      float e = s1[n] + s2[j];
      e = (e > 0.f) ? e : ALPHA * e;                    // leaky_relu
      e = (s_adj[n * NN + j] > 0.f) ? e : -9.0e15f;     // adjacency mask
      er[j] = e;
      mx = fmaxf(mx, e);
    }
    float sum = 0.f;
#pragma unroll
    for (int j = 0; j < NN; ++j) {
      float p = __expf(er[j] - mx);
      er[j] = p;
      sum += p;
    }
    float inv = 1.f / sum;
    float acc[4] = {0.f, 0.f, 0.f, 0.f};
#pragma unroll
    for (int j = 0; j < NN; ++j) {
      float wj = er[j] * inv;
#pragma unroll
      for (int q = 0; q < 4; ++q) acc[q] += wj * wh[j * HH + h0 + q];
    }
#pragma unroll
    for (int q = 0; q < 4; ++q) {
      float hv = acc[q] + br[q];
      hv = (hv > 0.f) ? hv : (__expf(hv) - 1.f);        // elu
      hpf[n * HH + h0 + q] = (__bf16)hv;
    }
  }
}

// ---------------------------------------------------------------------------
// Kernel 3: conv1d(k=3, pad=1) as 3 shifted GEMM passes.
// y1[m][o] = sum_tap sum_i hp[m+tap-1][i] * wk[tap][o][i]  (+conv_b)
// Block = 256 threads = 8 waves -> 128(M) x 64(N) tile.
// Each wave: one 16x64 strip = 4 accumulators sharing one A fragment.
// Out-of-range taps handled by a per-lane pointer select to a zero row,
// hoisted out of the K loop (branch-free inner loop).
// ---------------------------------------------------------------------------
__global__ __launch_bounds__(256) void conv_gemm(
    const __bf16* __restrict__ hp, const __bf16* __restrict__ wk,
    const __bf16* __restrict__ zrow, const float* __restrict__ conv_b,
    __bf16* __restrict__ y1) {
  const int lane  = threadIdx.x & 31;
  const int wave  = threadIdx.x >> 5;
  const int tileM = blockIdx.y * 128 + wave * 16;
  const int tileN = blockIdx.x * 64;

  const int mRow  = tileM + (lane & 15);
  const int aKb   = (lane >> 4) * 8;
  const int bKoff = (lane >> 4) * 8;
  const int n0    = tileN + (lane & 15);       // cols n0, n0+16, n0+32, n0+48
  const int t     = mRow & (TT - 1);

  v8f acc0 = {}, acc1 = {}, acc2 = {}, acc3 = {};

  for (int tap = 0; tap < 3; ++tap) {
    const int dt = tap - 1;
    const bool valid = ((unsigned)(t + dt) < (unsigned)TT);  // stay inside b
    const __bf16* aRow = valid ? (hp + (size_t)(mRow + dt) * NH) : zrow;
    const __bf16* bt   = wk + (size_t)tap * (OO * NH) + bKoff;
    const __bf16* b0   = bt + (size_t)(n0)      * NH;
    const __bf16* b1   = bt + (size_t)(n0 + 16) * NH;
    const __bf16* b2   = bt + (size_t)(n0 + 32) * NH;
    const __bf16* b3   = bt + (size_t)(n0 + 48) * NH;
#pragma unroll 2
    for (int ks = 0; ks < NH; ks += 32) {
      v16bf af  = load_a_frag(aRow + ks, aKb);
      v16bf bf0 = load_b_frag(b0 + ks);
      v16bf bf1 = load_b_frag(b1 + ks);
      v16bf bf2 = load_b_frag(b2 + ks);
      v16bf bf3 = load_b_frag(b3 + ks);
      acc0 = wmma_bf16(af, bf0, acc0);
      acc1 = wmma_bf16(af, bf1, acc1);
      acc2 = wmma_bf16(af, bf2, acc2);
      acc3 = wmma_bf16(af, bf3, acc3);
    }
  }

  const float cb0 = conv_b[n0];
  const float cb1 = conv_b[n0 + 16];
  const float cb2 = conv_b[n0 + 32];
  const float cb3 = conv_b[n0 + 48];
  const int rowBase = tileM + (lane >> 4) * 8;
#pragma unroll
  for (int r = 0; r < 8; ++r) {
    const size_t m = (size_t)(rowBase + r);
    y1[m * OO + n0]      = (__bf16)(acc0[r] + cb0);
    y1[m * OO + n0 + 16] = (__bf16)(acc1[r] + cb1);
    y1[m * OO + n0 + 32] = (__bf16)(acc2[r] + cb2);
    y1[m * OO + n0 + 48] = (__bf16)(acc3[r] + cb3);
  }
}

// ---------------------------------------------------------------------------
// Kernel 4: projection GEMM (K=256) + bias + mask -> f32 output
// out[m][o] = (sum_j y1[m][j] * proj_w[o][j] + proj_b[o]) * mask[m]
// Same 16x64-per-wave structure, K fully unrolled (8 steps).
// ---------------------------------------------------------------------------
__global__ __launch_bounds__(256) void proj_gemm(
    const __bf16* __restrict__ y1, const __bf16* __restrict__ pw,
    const float* __restrict__ proj_b, const float* __restrict__ mask,
    float* __restrict__ out) {
  const int lane  = threadIdx.x & 31;
  const int wave  = threadIdx.x >> 5;
  const int tileM = blockIdx.y * 128 + wave * 16;
  const int tileN = blockIdx.x * 64;

  const int mRow  = tileM + (lane & 15);
  const int aKb   = (lane >> 4) * 8;
  const int bKoff = (lane >> 4) * 8;
  const int n0    = tileN + (lane & 15);

  const __bf16* aRow = y1 + (size_t)mRow * OO;
  const __bf16* b0   = pw + (size_t)(n0)      * OO + bKoff;
  const __bf16* b1   = pw + (size_t)(n0 + 16) * OO + bKoff;
  const __bf16* b2   = pw + (size_t)(n0 + 32) * OO + bKoff;
  const __bf16* b3   = pw + (size_t)(n0 + 48) * OO + bKoff;

  v8f acc0 = {}, acc1 = {}, acc2 = {}, acc3 = {};
#pragma unroll
  for (int ks = 0; ks < OO; ks += 32) {
    v16bf af  = load_a_frag(aRow + ks, aKb);
    v16bf bf0 = load_b_frag(b0 + ks);
    v16bf bf1 = load_b_frag(b1 + ks);
    v16bf bf2 = load_b_frag(b2 + ks);
    v16bf bf3 = load_b_frag(b3 + ks);
    acc0 = wmma_bf16(af, bf0, acc0);
    acc1 = wmma_bf16(af, bf1, acc1);
    acc2 = wmma_bf16(af, bf2, acc2);
    acc3 = wmma_bf16(af, bf3, acc3);
  }

  const float pb0 = proj_b[n0];
  const float pb1 = proj_b[n0 + 16];
  const float pb2 = proj_b[n0 + 32];
  const float pb3 = proj_b[n0 + 48];
  const int rowBase = tileM + (lane >> 4) * 8;
#pragma unroll
  for (int r = 0; r < 8; ++r) {
    const int m = rowBase + r;
    const float mk = mask[m];
    float* o = out + (size_t)m * OO + n0;
    o[0]  = (acc0[r] + pb0) * mk;
    o[16] = (acc1[r] + pb1) * mk;
    o[32] = (acc2[r] + pb2) * mk;
    o[48] = (acc3[r] + pb3) * mk;
  }
}

// ---------------------------------------------------------------------------
// Launch
// ---------------------------------------------------------------------------
extern "C" void kernel_launch(void* const* d_in, const int* in_sizes, int n_in,
                              void* d_out, int out_size, void* d_ws, size_t ws_size,
                              hipStream_t stream) {
  const float* x      = (const float*)d_in[0];  // (B,T,N,C)
  const float* adj    = (const float*)d_in[1];  // (N,N)
  const float* mask   = (const float*)d_in[2];  // (B,T)
  const float* W      = (const float*)d_in[3];  // (C,H)
  const float* a      = (const float*)d_in[4];  // (2H,1)
  const float* gbias  = (const float*)d_in[5];  // (H,)
  const float* conv_w = (const float*)d_in[6];  // (O, N*H, 3)
  const float* conv_b = (const float*)d_in[7];  // (O,)
  const float* proj_w = (const float*)d_in[8];  // (O, O)
  const float* proj_b = (const float*)d_in[9];  // (O,)
  float* out = (float*)d_out;

  char* ws = (char*)d_ws;
  // 256B-aligned workspace carve-up (total ~96.1 MB)
  __bf16* hp   = (__bf16*)(ws);                 // 16384*2688*2 = 88080384
  __bf16* wk   = (__bf16*)(ws + 88080384);      // 3*256*2688*2 = 4128768
  __bf16* pw   = (__bf16*)(ws + 92209152);      // 256*256*2    = 131072
  __bf16* y1   = (__bf16*)(ws + 92340224);      // 16384*256*2  = 8388608
  __bf16* zrow = (__bf16*)(ws + 100728832);     // 2688*2 (zero pad row)

  // 1) weight repack to bf16 + zero-pad row init
  {
    const int total = 3 * OO * NH + OO * OO + NH;   // 2132608
    convert_weights<<<(total + 255) / 256, 256, 0, stream>>>(conv_w, proj_w, wk,
                                                             pw, zrow);
  }
  // 2) GAT: 16384 frames, 4 frames (waves) per 128-thread block
  gat_kernel<<<FF / 4, 128, 0, stream>>>(x, adj, W, a, gbias, hp);
  // 3) conv as GEMM: grid = (N/64, M/128) = (4, 128)
  {
    dim3 grid(OO / 64, FF / 128);
    conv_gemm<<<grid, 256, 0, stream>>>(hp, wk, zrow, conv_b, y1);
  }
  // 4) projection GEMM + mask
  {
    dim3 grid(OO / 64, FF / 128);
    proj_gemm<<<grid, 256, 0, stream>>>(y1, pw, proj_b, mask, out);
  }
}
